// RotatedBasicConv_77747497992673
// MI455X (gfx1250) — matrix operand order repaired and checked by reading, not compiled, not run on hardware
//
#include <hip/hip_runtime.h>
#include <hip/hip_fp16.h>

typedef __attribute__((ext_vector_type(16))) _Float16 v16h;
typedef __attribute__((ext_vector_type(8)))  _Float16 v8h;
typedef __attribute__((ext_vector_type(8)))  float    v8f;

#define NB     16        // batch
#define CIN    256
#define COUT   256
#define HWPIX  4096      // 64*64
#define KTOT   2304      // 9*CIN, K order: r*256+ci (filter tap outer, ci inner)
#define LDSPAD 40        // padded inner stride in halfs (80B: aligned, conflict-free)
#define NSLAB  8         // 8 ci-slabs of 32 channels
#define HELEM  (4 * 66 * 32)   // halo tile elements = 8448 = 33*256

// ------------------------------------------------------------------
// Kernel 1: global average pool  pooled[b*CIN+c] = mean(x[b,c,:,:])
// ------------------------------------------------------------------
__global__ void k_gap(const float* __restrict__ x, float* __restrict__ pooled)
{
  const float* p = x + (size_t)blockIdx.x * HWPIX;
  float s = 0.f;
  for (int i = threadIdx.x; i < HWPIX; i += 256) s += p[i];
  #pragma unroll
  for (int o = 16; o > 0; o >>= 1) s += __shfl_down(s, o, 32);
  __shared__ float red[8];
  if ((threadIdx.x & 31) == 0) red[threadIdx.x >> 5] = s;
  __syncthreads();
  if (threadIdx.x == 0) {
    float t = 0.f;
    #pragma unroll
    for (int i = 0; i < 8; ++i) t += red[i];
    pooled[blockIdx.x] = t * (1.0f / HWPIX);
  }
}

// ------------------------------------------------------------------
// Kernel 2: routing heads + scaled rotation matrices + folded BN.
// rotArr is stored pre-transposed as [b*9+i][n*9+j] (144 x 36).
// ------------------------------------------------------------------
__global__ void k_route(const float* __restrict__ pooled,
                        const float* __restrict__ w_lambda,
                        const float* __restrict__ w_theta,
                        const float* __restrict__ bn_gamma,
                        const float* __restrict__ bn_beta,
                        const float* __restrict__ bn_mean,
                        const float* __restrict__ bn_var,
                        float* __restrict__ rotArr,
                        float* __restrict__ bnscale,
                        float* __restrict__ bnshift)
{
  const int t = threadIdx.x;
  { // fold BN: y = o*scale + shift
    float iv = rsqrtf(bn_var[t] + 1e-5f);
    float sc = bn_gamma[t] * iv;
    bnscale[t] = sc;
    bnshift[t] = bn_beta[t] - bn_mean[t] * sc;
  }
  if (t < NB * 4) {
    const int b = t >> 2, n = t & 3;
    float dl = 0.f, dt = 0.f;
    for (int c = 0; c < CIN; ++c) {
      float pv = pooled[b * CIN + c];
      dl += pv * w_lambda[c * 4 + n];
      dt += pv * w_theta[c * 4 + n];
    }
    const float lam = 1.f / (1.f + expf(-dl));
    const float th  = (dt / (1.f + fabsf(dt))) * 3.14159265358979323846f;
    const float cx = cosf(th), sy = sinf(th);
    const float a = cx - sy, bxy = cx * sy, c = cx + sy;
    float R[81];
    #pragma unroll
    for (int i = 0; i < 81; ++i) R[i] = 0.f;
    if (th >= 0.f) {
      R[ 0]=a;            R[ 1]=1.f-a;
      R[ 9+1]=cx-bxy;     R[ 9+2]=bxy;       R[ 9+4]=1.f-c+bxy; R[ 9+5]=sy-bxy;
      R[18+2]=a;          R[18+5]=1.f-a;
      R[27+0]=bxy;        R[27+1]=sy-bxy;    R[27+3]=cx-bxy;    R[27+4]=1.f-c+bxy;
      R[36+4]=1.f;
      R[45+4]=1.f-c+bxy;  R[45+5]=cx-bxy;    R[45+7]=sy-bxy;    R[45+8]=bxy;
      R[54+3]=1.f-a;      R[54+6]=a;
      R[63+3]=sy-bxy;     R[63+4]=1.f-c+bxy; R[63+6]=bxy;       R[63+7]=cx-bxy;
      R[72+7]=1.f-a;      R[72+8]=a;
    } else {
      R[ 0]=c;            R[ 3]=1.f-c;
      R[ 9+0]=-bxy;       R[ 9+1]=cx+bxy;    R[ 9+3]=bxy-sy;    R[ 9+4]=1.f-a-bxy;
      R[18+1]=1.f-c;      R[18+2]=c;
      R[27+3]=cx+bxy;     R[27+4]=1.f-a-bxy; R[27+6]=-bxy;      R[27+7]=bxy-sy;
      R[36+4]=1.f;
      R[45+1]=bxy-sy;     R[45+2]=-bxy;      R[45+4]=1.f-a-bxy; R[45+5]=cx+bxy;
      R[54+6]=c;          R[54+7]=1.f-c;
      R[63+4]=1.f-a-bxy;  R[63+5]=bxy-sy;    R[63+7]=cx+bxy;    R[63+8]=-bxy;
      R[72+5]=1.f-c;      R[72+8]=c;
    }
    for (int i = 0; i < 9; ++i)
      for (int j = 0; j < 9; ++j)
        rotArr[(b * 9 + i) * 36 + n * 9 + j] = lam * R[i * 9 + j];
  }
}

// ------------------------------------------------------------------
// Kernel 3: rotate/mix weights -> f16, GEMM layout rw[b*COUT+co][r*256+ci]
// rw[...][i*256+ci] = sum_{n,j} rotArr[b*9+i][n*9+j]*W[n,co,ci,j]
// ------------------------------------------------------------------
__global__ void k_rotw(const float* __restrict__ weight,
                       const float* __restrict__ rotArr,
                       _Float16* __restrict__ rw)
{
  const int co = blockIdx.x;     // 0..255
  const int b  = blockIdx.y;     // 0..15
  __shared__ float rloc[9 * 36];
  for (int i = threadIdx.x; i < 324; i += 256) rloc[i] = rotArr[b * 9 * 36 + i];
  __syncthreads();
  const int ci = threadIdx.x;    // 0..255
  float wv[36];
  const float* wp = weight + ((size_t)co * CIN + ci) * 9;   // + n*COUT*CIN*9
  #pragma unroll
  for (int n = 0; n < 4; ++n)
    #pragma unroll
    for (int j = 0; j < 9; ++j)
      wv[n * 9 + j] = wp[(size_t)n * COUT * CIN * 9 + j];
  _Float16* out = rw + (size_t)(b * COUT + co) * KTOT + ci;
  #pragma unroll
  for (int i = 0; i < 9; ++i) {
    float acc = 0.f;
    #pragma unroll
    for (int q = 0; q < 36; ++q) acc += rloc[i * 36 + q] * wv[q];
    out[i * CIN] = (_Float16)acc;           // tap-major K: k = i*256 + ci
  }
}

// ------------------------------------------------------------------
// Kernel 4: implicit-GEMM conv + BN + ReLU via v_wmma_f32_16x16x32_f16
// per batch: M=COUT=256, N=4096 pixels, K=2304 (tap-major). Block tile
// 128x128. K loop = 8 ci-slabs x 9 taps; one LDS halo tile per slab is
// reused by all 9 taps (x gathered once instead of 9x). A tiles are
// double-buffered per tap. 8 waves of 64x32 (4x2 WMMA accumulators).
// ------------------------------------------------------------------
__device__ __forceinline__ void load_halo(const float* __restrict__ xB,
                                          int slab, int h0,
                                          _Float16 (*xh)[66][LDSPAD], int tid)
{
  const int ci0 = slab * 32;
  // 4 rows x 66 cols x 32 ci, col fastest (coalesced), ci slowest
  #pragma unroll 1
  for (int e = tid; e < HELEM; e += 256) {
    const int col = e % 66;
    const int t   = e / 66;          // ci*4 + row
    const int row = t & 3;
    const int ci  = t >> 2;
    const int hh  = h0 + row - 1;
    const int ww  = col - 1;
    float v = 0.f;
    if ((unsigned)hh < 64u && (unsigned)ww < 64u)
      v = xB[(size_t)(ci0 + ci) * HWPIX + hh * 64 + ww];
    xh[row][col][ci] = (_Float16)v;
  }
}

__device__ __forceinline__ void load_A(const _Float16* __restrict__ rwB,
                                       int slab, int tap,
                                       _Float16 (*as)[LDSPAD], int tid)
{
  const int kc = tap * CIN + slab * 32;   // tap-major K offset
  #pragma unroll
  for (int i = 0; i < 2; ++i) {
    int s  = tid + i * 256;          // 0..511
    int m  = s >> 2;                 // 0..127
    int k8 = (s & 3) * 8;            // 0/8/16/24
    v8h val = *(const v8h*)(rwB + (size_t)m * KTOT + kc + k8);
    *(v8h*)&as[m][k8] = val;
  }
}

__device__ __forceinline__ void mma_tap(const _Float16 (*as)[LDSPAD],
                                        const _Float16 (*xh)[66][LDSPAD],
                                        v8f acc[4][2], int wm, int wn,
                                        int lane, int kh, int kw)
{
  const int l16 = lane & 15;
  const int kbA = (lane >> 4) * 8;    // A: K pairs 0-7/16-23 vs 8-15/24-31
  const int kbB = (lane >> 4) * 8;    // B: K 0-7/16-23 vs 8-15/24-31 (matches A)
  v16h af[4], bf[2];
  #pragma unroll
  for (int i = 0; i < 4; ++i) {
    const _Float16* rp = &as[wm + i * 16 + l16][0];
    v8h lo = *(const v8h*)(rp + kbA);
    v8h hi = *(const v8h*)(rp + kbA + 16);
    af[i] = __builtin_shufflevector(lo, hi, 0,1,2,3,4,5,6,7,8,9,10,11,12,13,14,15);
  }
  #pragma unroll
  for (int j = 0; j < 2; ++j) {
    const int n  = wn + j * 16 + l16;     // pixel in 128-tile
    const int hr = (n >> 6) + kh;         // halo row  (0..3)
    const int wc = (n & 63) + kw;         // halo col  (0..65)
    const _Float16* rp = &xh[hr][wc][0];
    v8h lo = *(const v8h*)(rp + kbB);
    v8h hi = *(const v8h*)(rp + kbB + 16);
    bf[j] = __builtin_shufflevector(lo, hi, 0,1,2,3,4,5,6,7,8,9,10,11,12,13,14,15);
  }
  #pragma unroll
  for (int i = 0; i < 4; ++i)
    #pragma unroll
    for (int j = 0; j < 2; ++j)
      acc[i][j] = __builtin_amdgcn_wmma_f32_16x16x32_f16(
          false, af[i], false, bf[j], (short)0, acc[i][j], false, false);
}

__global__ __launch_bounds__(256)
void k_conv(const float* __restrict__ x, const _Float16* __restrict__ rw,
            const float* __restrict__ bnscale, const float* __restrict__ bnshift,
            float* __restrict__ out)
{
  __shared__ _Float16 as[2][128][LDSPAD];   // 20.0 KiB (double-buffered A)
  __shared__ _Float16 xh[4][66][LDSPAD];    // 20.6 KiB (halo tile, ci inner)
  const int tid   = threadIdx.x;
  const int b     = blockIdx.z;
  const int mBase = blockIdx.y * 128;       // co tile
  const int pBase = blockIdx.x * 128;       // pixel tile (2 image rows)
  const int h0    = blockIdx.x * 2;
  const _Float16* rwB = rw + (size_t)(b * COUT + mBase) * KTOT;
  const float*    xB  = x  + (size_t)b * CIN * HWPIX;

  const int lane = tid & 31;
  const int wv   = tid >> 5;                // 0..7
  const int wm   = (wv & 1) * 64;           // wave tile 64(M) x 32(N)
  const int wn   = (wv >> 1) * 32;

  v8f acc[4][2];
  const v8f zero = {0.f,0.f,0.f,0.f,0.f,0.f,0.f,0.f};
  #pragma unroll
  for (int i = 0; i < 4; ++i)
    #pragma unroll
    for (int j = 0; j < 2; ++j) acc[i][j] = zero;

  #pragma unroll 1
  for (int slab = 0; slab < NSLAB; ++slab) {
    load_halo(xB, slab, h0, xh, tid);       // x gathered once per slab
    load_A(rwB, slab, 0, as[0], tid);
    __syncthreads();                        // halo + A(tap0) visible
    #pragma unroll 1
    for (int tap = 0; tap < 9; ++tap) {
      if (tap < 8)
        load_A(rwB, slab, tap + 1, as[(tap + 1) & 1], tid);
      if (tap == 0 && slab < NSLAB - 1) {   // warm L2: next slab's x rows
        int ci = (slab + 1) * 32 + (tid >> 3);
        int rr = min(max(h0 - 1 + (tid & 7), 0), 63);
        __builtin_prefetch(xB + (size_t)ci * HWPIX + rr * 64, 0, 1);
      }
      mma_tap(as[tap & 1], xh, acc, wm, wn, lane, tap / 3, tap % 3);
      __syncthreads();
    }
  }

  // epilogue: C layout -> lane l16 = N col, vgpr v + 8*(lane>>4) = M row
  const int l16  = lane & 15;
  const int half = lane >> 4;
  #pragma unroll
  for (int i = 0; i < 4; ++i) {
    #pragma unroll
    for (int v = 0; v < 8; ++v) {
      const int co = mBase + wm + i * 16 + half * 8 + v;
      const float sc = bnscale[co];
      const float sh = bnshift[co];
      float* op = out + ((size_t)(b * COUT + co)) * HWPIX + pBase;
      #pragma unroll
      for (int j = 0; j < 2; ++j) {
        float val = acc[i][j][v] * sc + sh;
        op[wn + j * 16 + l16] = fmaxf(val, 0.f);
      }
    }
  }
}

// ------------------------------------------------------------------
extern "C" void kernel_launch(void* const* d_in, const int* in_sizes, int n_in,
                              void* d_out, int out_size, void* d_ws, size_t ws_size,
                              hipStream_t stream)
{
  (void)in_sizes; (void)n_in; (void)out_size; (void)ws_size;
  const float* x        = (const float*)d_in[0];
  const float* weight   = (const float*)d_in[1];
  const float* w_lambda = (const float*)d_in[2];
  const float* w_theta  = (const float*)d_in[3];
  const float* bn_gamma = (const float*)d_in[4];
  const float* bn_beta  = (const float*)d_in[5];
  const float* bn_mean  = (const float*)d_in[6];
  const float* bn_var   = (const float*)d_in[7];
  float* out = (float*)d_out;

  char* ws = (char*)d_ws;
  float*    pooled  = (float*)(ws);                          // 16384 B
  float*    rotArr  = (float*)(ws + 16384);                  // 20736 B
  float*    bnscale = (float*)(ws + 16384 + 20736);          // 1024 B
  float*    bnshift = (float*)(ws + 16384 + 20736 + 1024);   // 1024 B
  _Float16* rw      = (_Float16*)(ws + 40960);               // 18,874,368 B

  k_gap  <<<NB * CIN, 256, 0, stream>>>(x, pooled);
  k_route<<<1, 256, 0, stream>>>(pooled, w_lambda, w_theta, bn_gamma, bn_beta,
                                 bn_mean, bn_var, rotArr, bnscale, bnshift);
  k_rotw <<<dim3(COUT, NB), 256, 0, stream>>>(weight, rotArr, rw);
  k_conv <<<dim3(HWPIX / 128, COUT / 128, NB), 256, 0, stream>>>(x, rw, bnscale,
                                                                 bnshift, out);
}
